// FlashAttentionV2_71451075936541
// MI455X (gfx1250) — matrix operand order repaired and checked by reading, not compiled, not run on hardware
//
#include <hip/hip_runtime.h>
#include <hip/hip_bf16.h>

// ---------------------------------------------------------------------------
// MHA layer for MI455X (gfx1250, wave32):
//   qkv = x @ w_qkv.T  ->  flash-attention (block=128)  ->  out @ w_out.T + b
// All GEMMs run on V_WMMA_F32_16X16X32_BF16 (bf16 in, f32 accumulate).
// GEMM uses a ping-pong LDS double buffer (64-wide K slab, 1 barrier/slab),
// filled via GLOBAL_LOAD_ASYNC_TO_LDS_B128 (ASYNCcnt-tracked).
// ---------------------------------------------------------------------------

typedef __attribute__((ext_vector_type(16))) __bf16       v16bf;
typedef __attribute__((ext_vector_type(8)))  float        v8f;
typedef __attribute__((ext_vector_type(4)))  unsigned int u32x4;
typedef int v4i_ __attribute__((__vector_size__(16)));    // matches builtin proto

#define E_DIM   2048
#define H_NUM   16
#define D_HEAD  128
#define S_LEN   4096
#define B_NUM   2
#define T_TOK   (B_NUM * S_LEN)   // 8192 tokens
#define QKV_N   (3 * E_DIM)       // 6144

#if defined(__gfx1250__) && __has_builtin(__builtin_amdgcn_global_load_async_to_lds_b128)
#define USE_ASYNC_LDS 1
#else
#define USE_ASYNC_LDS 0
#endif

__device__ __forceinline__ void wait_async_lds() {
#if USE_ASYNC_LDS
#if __has_builtin(__builtin_amdgcn_s_wait_asynccnt)
  __builtin_amdgcn_s_wait_asynccnt(0);
#else
  asm volatile("s_wait_asynccnt 0x0" ::: "memory");
#endif
#endif
}

// Copy NB*16 bytes (NB b128 beats) from global to LDS for this lane.
template <int NB>
__device__ __forceinline__ void glds_copy(const __bf16* __restrict__ g,
                                          __bf16* __restrict__ l) {
#if USE_ASYNC_LDS
#pragma unroll
  for (int i = 0; i < NB; ++i) {
    __builtin_amdgcn_global_load_async_to_lds_b128(
        (__attribute__((address_space(1))) v4i_*)(g + i * 8),
        (__attribute__((address_space(3))) v4i_*)(l + i * 8), 0, 0);
  }
#else
  u32x4 t[NB];
#pragma unroll
  for (int i = 0; i < NB; ++i) t[i] = *(const u32x4*)(g + i * 8);
#pragma unroll
  for (int i = 0; i < NB; ++i) *(u32x4*)(l + i * 8) = t[i];
#endif
}

__device__ __forceinline__ __bf16 f2bf(float f) {
  union { float f; unsigned u; } a; a.f = f;
  unsigned r = a.u + 0x7FFFu + ((a.u >> 16) & 1u);  // round-to-nearest-even
  union { unsigned short s; __bf16 b; } o;
  o.s = (unsigned short)(r >> 16);
  return o.b;
}

union Frag16 { v16bf v; u32x4 q[2]; };

// ---------------------------------------------------------------------------
// f32 -> bf16 bulk convert (vectorized x4)
// ---------------------------------------------------------------------------
__global__ __launch_bounds__(256)
void cvt_f32_bf16(const float* __restrict__ in, __bf16* __restrict__ out, int n4) {
  int i = blockIdx.x * blockDim.x + threadIdx.x;
  if (i >= n4) return;
  float4 f = ((const float4*)in)[i];
  union { __bf16 h[4]; unsigned long long u; } o;
  o.h[0] = f2bf(f.x); o.h[1] = f2bf(f.y); o.h[2] = f2bf(f.z); o.h[3] = f2bf(f.w);
  ((unsigned long long*)out)[i] = o.u;
}

// ---------------------------------------------------------------------------
// C[M,N] = A[M,K] * B[N,K]^T  (both K-contiguous, bf16), 128x128 tile / block.
// 8 waves; wave computes 32x64 (2x4 of 16x16 WMMA tiles). K slab = 64,
// LDS ping-pong double buffer, one barrier per slab.
// mode 0: scatter bf16 into q/k/v [B,H,S,D]   (QKV projection, N = 6144)
// mode 1: f32 out[t*E + n] = acc + bias[n]    (output projection, N = 2048)
// ---------------------------------------------------------------------------
#define GKS  64   // K slab
#define GSTR 72   // LDS row stride in bf16 (64 + 8 pad -> conflict-free b128)
#define GEMM_LDS_BYTES ((size_t)4 * 128 * GSTR * sizeof(__bf16))  // 73728

__global__ __launch_bounds__(256)
void gemm_bf16(const __bf16* __restrict__ A, const __bf16* __restrict__ Bm,
               int K, int mode,
               __bf16* __restrict__ qout, __bf16* __restrict__ kout,
               __bf16* __restrict__ vout,
               float* __restrict__ outf, const float* __restrict__ bias) {
  extern __shared__ __align__(16) char gsmem[];
  __bf16* As = (__bf16*)gsmem;             // 2 buffers x 128 x GSTR
  __bf16* Bs = As + 2 * 128 * GSTR;        // 2 buffers x 128 x GSTR

  const int tid  = threadIdx.x;
  const int lane = tid & 31;
  const int wave = tid >> 5;
  const int hi16 = lane >> 4;      // lane half (C-layout row offset 0/8)
  const int l16  = lane & 15;
  const int m0 = blockIdx.y * 128, n0 = blockIdx.x * 128;
  const int mw = (wave >> 1) * 32, nw = (wave & 1) * 64;
  const int lrow = tid >> 1, lseg = (tid & 1) * 32;   // 2 threads/row, 32 cols

  v8f acc[2][4];
#pragma unroll
  for (int mi = 0; mi < 2; ++mi) {
#pragma unroll
    for (int ni = 0; ni < 4; ++ni) {
#pragma unroll
      for (int r = 0; r < 8; ++r) acc[mi][ni][r] = 0.0f;
    }
  }

  const __bf16* Ag = A  + (size_t)(m0 + lrow) * K + lseg;
  const __bf16* Bg = Bm + (size_t)(n0 + lrow) * K + lseg;
  const int loff = lrow * GSTR + lseg;

  // Prologue: fill buffer 0 with slab 0.
  glds_copy<4>(Ag, As + loff);
  glds_copy<4>(Bg, Bs + loff);
  wait_async_lds();
  __syncthreads();

  int buf = 0;
  for (int k0 = 0; k0 < K; k0 += GKS, buf ^= 1) {
    // Kick off next slab into the other buffer (overlaps with WMMAs below).
    if (k0 + GKS < K) {
      const int nb = (buf ^ 1) * 128 * GSTR;
      __builtin_prefetch(Ag + k0 + 2 * GKS, 0, 3);   // global_prefetch_b8
      __builtin_prefetch(Bg + k0 + 2 * GKS, 0, 3);
      glds_copy<4>(Ag + k0 + GKS, As + nb + loff);
      glds_copy<4>(Bg + k0 + GKS, Bs + nb + loff);
    }

    const __bf16* Ab = As + buf * 128 * GSTR;
    const __bf16* Bb = Bs + buf * 128 * GSTR;
#pragma unroll
    for (int kk = 0; kk < 2; ++kk) {
      Frag16 fa[2], fb[4];
#pragma unroll
      for (int mi = 0; mi < 2; ++mi) {
        // ISA 16-bit A layout: K in {0..7,16..23} (+8 for lanes 16..31)
        const __bf16* p = &Ab[(mw + mi * 16 + l16) * GSTR + kk * 32 + hi16 * 8];
        fa[mi].q[0] = *(const u32x4*)p;
        fa[mi].q[1] = *(const u32x4*)(p + 16);
      }
#pragma unroll
      for (int ni = 0; ni < 4; ++ni) {
        // B layout: lane = column n, K in {0..7,16..23} (+8 for lanes 16..31)
        const __bf16* p = &Bb[(nw + ni * 16 + l16) * GSTR + kk * 32 + hi16 * 8];
        fb[ni].q[0] = *(const u32x4*)p;
        fb[ni].q[1] = *(const u32x4*)(p + 16);
      }
#pragma unroll
      for (int mi = 0; mi < 2; ++mi) {
#pragma unroll
        for (int ni = 0; ni < 4; ++ni) {
          acc[mi][ni] = __builtin_amdgcn_wmma_f32_16x16x32_bf16(
              false, fa[mi].v, false, fb[ni].v, (short)0, acc[mi][ni],
              false, false);
        }
      }
    }
    wait_async_lds();   // own async fills of the next buffer landed
    __syncthreads();    // everyone done reading cur + writes to nxt visible
  }

  // Epilogue. C layout: row = r + 8*hi16, col = l16.
#pragma unroll
  for (int mi = 0; mi < 2; ++mi) {
#pragma unroll
    for (int ni = 0; ni < 4; ++ni) {
#pragma unroll
      for (int r = 0; r < 8; ++r) {
        int m = m0 + mw + mi * 16 + hi16 * 8 + r;   // token t
        int n = n0 + nw + ni * 16 + l16;
        float val = acc[mi][ni][r];
        if (mode == 0) {
          int c = n >> 11, rem = n & 2047;          // qkv split
          int h = rem >> 7, d = rem & 127;
          int b = m >> 12, s = m & 4095;
          __bf16* dst = (c == 0) ? qout : (c == 1) ? kout : vout;
          dst[(((size_t)b * H_NUM + h) * S_LEN + s) * D_HEAD + d] = f2bf(val);
        } else {
          outf[(size_t)m * E_DIM + n] = val + bias[n];
        }
      }
    }
  }
}

// ---------------------------------------------------------------------------
// FlashAttention-v2 block kernel, faithful to the reference rescaling
// (including the extra l_ij factor):
//   out = (alpha*l/l_new)*out + (beta*l_ij/l_new)*(P @ V)
// One workgroup per (b, h, q-block of 128). 8 waves; wave owns 16 q rows.
// LDS: Qs (resident), KPs (K tile, reused for bf16 P), Vt (V transposed).
// Row stride 136 bf16 = 272B -> rotates 4 banks/row (conflict-free frags).
// ---------------------------------------------------------------------------
#define SSTR 136

__global__ __launch_bounds__(256)
void flash_attn(const __bf16* __restrict__ Q, const __bf16* __restrict__ Kmat,
                const __bf16* __restrict__ V, __bf16* __restrict__ O) {
  extern __shared__ __align__(16) char smem_raw[];
  __bf16* Qs  = (__bf16*)smem_raw;       // 128 x SSTR
  __bf16* KPs = Qs  + 128 * SSTR;        // K tile, later P tile
  __bf16* Vt  = KPs + 128 * SSTR;        // V transposed: [d][key]

  const int tid  = threadIdx.x;
  const int lane = tid & 31;
  const int wave = tid >> 5;
  const int hi16 = lane >> 4;
  const int l16  = lane & 15;
  const int qw   = wave * 16;            // wave's first q row in block

  const int bid  = blockIdx.x;
  const int qblk = bid & 31;
  const int h    = (bid >> 5) & 15;
  const int b    = bid >> 9;

  const size_t bh = ((size_t)b * H_NUM + h) * S_LEN;
  const __bf16* Qg = Q    + (bh + (size_t)qblk * 128) * D_HEAD;
  const __bf16* Kg = Kmat + bh * D_HEAD;
  const __bf16* Vg = V    + bh * D_HEAD;

  const int row = tid >> 1, seg = (tid & 1) * 64;   // tile-loader mapping

  // Load Q tile (128x128 bf16); first use is after the in-loop wait+barrier.
  glds_copy<8>(Qg + (size_t)row * D_HEAD + seg, Qs + row * SSTR + seg);

  float m_run[8], l_run[8];
  v8f oacc[8];
#pragma unroll
  for (int r = 0; r < 8; ++r) { m_run[r] = -INFINITY; l_run[r] = 0.0f; }
#pragma unroll
  for (int t = 0; t < 8; ++t) {
#pragma unroll
    for (int r = 0; r < 8; ++r) oacc[t][r] = 0.0f;
  }

  const float scale = 0.08838834764831845f;   // 1/sqrt(128)

  for (int j = 0; j < S_LEN / 128; ++j) {
    __syncthreads();   // prior iteration's P/V reads complete
    // K tile: straight b128 copy (async path when available)
    glds_copy<8>(Kg + (size_t)(j * 128 + row) * D_HEAD + seg,
                 KPs + row * SSTR + seg);
    // V tile: transpose on the way in so PV B-fragments are contiguous
    {
      const u32x4* vs =
          (const u32x4*)(Vg + (size_t)(j * 128 + row) * D_HEAD + seg);
#pragma unroll
      for (int i = 0; i < 8; ++i) {
        union { u32x4 q; __bf16 e[8]; } tmp;
        tmp.q = vs[i];
#pragma unroll
        for (int e = 0; e < 8; ++e)
          Vt[(seg + i * 8 + e) * SSTR + row] = tmp.e[e];
      }
    }
    wait_async_lds();
    __syncthreads();

    // ---- S = Q * K^T  (wave's 16 rows x all 128 keys) ----
    v8f sacc[8];
#pragma unroll
    for (int t = 0; t < 8; ++t) {
#pragma unroll
      for (int r = 0; r < 8; ++r) sacc[t][r] = 0.0f;
    }
#pragma unroll
    for (int kk = 0; kk < 4; ++kk) {
      Frag16 fa;
      const __bf16* ap = &Qs[(qw + l16) * SSTR + kk * 32 + hi16 * 8];
      fa.q[0] = *(const u32x4*)ap;
      fa.q[1] = *(const u32x4*)(ap + 16);
#pragma unroll
      for (int nt = 0; nt < 8; ++nt) {
        Frag16 fb;
        const __bf16* bp = &KPs[(nt * 16 + l16) * SSTR + kk * 32 + hi16 * 8];
        fb.q[0] = *(const u32x4*)bp;
        fb.q[1] = *(const u32x4*)(bp + 16);
        sacc[nt] = __builtin_amdgcn_wmma_f32_16x16x32_bf16(
            false, fa.v, false, fb.v, (short)0, sacc[nt], false, false);
      }
    }

    // ---- online softmax (rows live as C-layout: row r+8*hi16) ----
    float mij[8], lij[8];
#pragma unroll
    for (int r = 0; r < 8; ++r) {
      float mx = sacc[0][r];
#pragma unroll
      for (int t = 1; t < 8; ++t) mx = fmaxf(mx, sacc[t][r]);
      mx *= scale;
      mx = fmaxf(mx, __shfl_xor(mx, 1));
      mx = fmaxf(mx, __shfl_xor(mx, 2));
      mx = fmaxf(mx, __shfl_xor(mx, 4));
      mx = fmaxf(mx, __shfl_xor(mx, 8));
      mij[r] = mx;
      lij[r] = 0.0f;
    }
#pragma unroll
    for (int t = 0; t < 8; ++t) {
#pragma unroll
      for (int r = 0; r < 8; ++r) {
        float p = __expf(sacc[t][r] * scale - mij[r]);
        sacc[t][r] = p;
        lij[r] += p;
      }
    }
#pragma unroll
    for (int r = 0; r < 8; ++r) {
      lij[r] += __shfl_xor(lij[r], 1);
      lij[r] += __shfl_xor(lij[r], 2);
      lij[r] += __shfl_xor(lij[r], 4);
      lij[r] += __shfl_xor(lij[r], 8);
    }

    __syncthreads();   // everyone is done reading K -> safe to overwrite w/ P
    // write P (bf16) into the wave's own 16 rows of KPs
#pragma unroll
    for (int t = 0; t < 8; ++t) {
#pragma unroll
      for (int r = 0; r < 8; ++r)
        KPs[(qw + hi16 * 8 + r) * SSTR + t * 16 + l16] = f2bf(sacc[t][r]);
    }

    // running stats + rescale of existing output (reference formula)
    float c1[8];
#pragma unroll
    for (int r = 0; r < 8; ++r) {
      float mn    = fmaxf(m_run[r], mij[r]);
      float alpha = __expf(m_run[r] - mn);
      float beta  = __expf(mij[r] - mn);
      float ln    = alpha * l_run[r] + beta * lij[r];
      float inv   = 1.0f / ln;
      float c0    = alpha * l_run[r] * inv;
      c1[r]       = beta * lij[r] * inv;     // extra l_ij, as in reference
      m_run[r] = mn; l_run[r] = ln;
#pragma unroll
      for (int t = 0; t < 8; ++t) oacc[t][r] *= c0;
    }

    // ---- PV: reuse sacc as the accumulator (saves 64 VGPRs) ----
#pragma unroll
    for (int t = 0; t < 8; ++t) {
#pragma unroll
      for (int r = 0; r < 8; ++r) sacc[t][r] = 0.0f;
    }
#pragma unroll
    for (int kk = 0; kk < 4; ++kk) {
      Frag16 fa;  // A = P, wave reads only its own rows (same-wave DS order)
      const __bf16* ap = &KPs[(qw + l16) * SSTR + kk * 32 + hi16 * 8];
      fa.q[0] = *(const u32x4*)ap;
      fa.q[1] = *(const u32x4*)(ap + 16);
#pragma unroll
      for (int dt = 0; dt < 8; ++dt) {
        Frag16 fb;  // B = V^T rows are d, contiguous over key
        const __bf16* bp = &Vt[(dt * 16 + l16) * SSTR + kk * 32 + hi16 * 8];
        fb.q[0] = *(const u32x4*)bp;
        fb.q[1] = *(const u32x4*)(bp + 16);
        sacc[dt] = __builtin_amdgcn_wmma_f32_16x16x32_bf16(
            false, fa.v, false, fb.v, (short)0, sacc[dt], false, false);
      }
    }
#pragma unroll
    for (int t = 0; t < 8; ++t) {
#pragma unroll
      for (int r = 0; r < 8; ++r) oacc[t][r] += c1[r] * sacc[t][r];
    }
  }

  // Store normalized output directly in [B,S,H*D] (token-major) bf16.
  __bf16* Og = O + ((size_t)b * S_LEN + (size_t)qblk * 128) * E_DIM
                 + (size_t)h * D_HEAD;
#pragma unroll
  for (int t = 0; t < 8; ++t) {
#pragma unroll
    for (int r = 0; r < 8; ++r) {
      int rw = qw + hi16 * 8 + r;
      Og[(size_t)rw * E_DIM + t * 16 + l16] = f2bf(oacc[t][r]);
    }
  }
}

// ---------------------------------------------------------------------------
extern "C" void kernel_launch(void* const* d_in, const int* in_sizes, int n_in,
                              void* d_out, int out_size, void* d_ws,
                              size_t ws_size, hipStream_t stream) {
  (void)in_sizes; (void)n_in; (void)out_size; (void)ws_size;
  const float* x     = (const float*)d_in[0];
  const float* w_qkv = (const float*)d_in[1];
  const float* w_out = (const float*)d_in[2];
  const float* b_out = (const float*)d_in[3];
  float* out = (float*)d_out;

  char* ws = (char*)d_ws;
  size_t off = 0;
  auto carve = [&](size_t bytes) -> char* {
    char* p = ws + off;
    off = (off + bytes + 255) & ~(size_t)255;
    return p;
  };
  __bf16* xb    = (__bf16*)carve((size_t)T_TOK * E_DIM * 2);   // 33.6 MB
  __bf16* wqkvb = (__bf16*)carve((size_t)QKV_N * E_DIM * 2);   // 25.2 MB
  __bf16* woutb = (__bf16*)carve((size_t)E_DIM * E_DIM * 2);   //  8.4 MB
  __bf16* qb    = (__bf16*)carve((size_t)T_TOK * E_DIM * 2);   // 33.6 MB
  __bf16* kb    = (__bf16*)carve((size_t)T_TOK * E_DIM * 2);
  __bf16* vb    = (__bf16*)carve((size_t)T_TOK * E_DIM * 2);
  __bf16* attnb = (__bf16*)carve((size_t)T_TOK * E_DIM * 2);   // total ~201 MB

  // 1) precision downcast (halves GEMM bandwidth; all operands L2-resident)
  int nx = T_TOK * E_DIM / 4, nw = QKV_N * E_DIM / 4, no = E_DIM * E_DIM / 4;
  cvt_f32_bf16<<<(nx + 255) / 256, 256, 0, stream>>>(x, xb, nx);
  cvt_f32_bf16<<<(nw + 255) / 256, 256, 0, stream>>>(w_qkv, wqkvb, nw);
  cvt_f32_bf16<<<(no + 255) / 256, 256, 0, stream>>>(w_out, woutb, no);

  // 2) QKV projection -> q/k/v in [B,H,S,D] bf16
  dim3 g1(QKV_N / 128, T_TOK / 128);
  gemm_bf16<<<g1, 256, GEMM_LDS_BYTES, stream>>>(xb, wqkvb, E_DIM, 0,
                                                 qb, kb, vb, nullptr, nullptr);

  // 3) flash attention, one block per (b,h,qblock); 3 LDS tiles of 128x136 bf16
  size_t flash_lds = (size_t)3 * 128 * SSTR * sizeof(__bf16);  // 104448 B
  flash_attn<<<B_NUM * H_NUM * (S_LEN / 128), 256, flash_lds, stream>>>(
      qb, kb, vb, attnb);

  // 4) output projection + bias -> f32
  dim3 g2(E_DIM / 128, T_TOK / 128);
  gemm_bf16<<<g2, 256, GEMM_LDS_BYTES, stream>>>(attnb, woutb, E_DIM, 1,
                                                 nullptr, nullptr, nullptr,
                                                 out, b_out);
}
